// GraphAttentionLayer_14551349198956
// MI455X (gfx1250) — matrix operand order repaired and checked by reading, not compiled, not run on hardware
//
#include <hip/hip_runtime.h>
#include <hip/hip_bf16.h>
#include <math.h>

// ---------------------------------------------------------------------------
// GAT forward, fused for MI455X (gfx1250, wave32, WMMA).
//   N = 8192, IN_F = 256, OUT_F = 128, alpha = 0.2
// Pipeline:
//   K1 : Wh = h @ W            (v_wmma_f32_16x16x4_f32), emits WhT bf16
//   K2 : wh1/wh2 = Wh @ a1/a2  (+ per-block max of wh2), pre-scaled by log2e
//   K2b: reduce block maxes -> wh2max scalar
//   K3 : single pass over adj (HBM floor: 268MB @ 23.3TB/s ~ 11.5us):
//        p = exp2(lrelu2(w1+w2) - c) masked; acc += P @ Wh and
//        accsum += P @ ones via v_wmma_f32_16x16x32_bf16; out = elu(acc/accsum)
// c_i = lrelu(wh1_i + max wh2) >= row max (lrelu monotone) => no online softmax.
// ---------------------------------------------------------------------------

#define GN    8192
#define INF_  256
#define OUTF  128
#define ALPHA 0.2f
#define LOG2E 1.44269504088896340736f

typedef __attribute__((ext_vector_type(2)))  float  v2f;
typedef __attribute__((ext_vector_type(8)))  float  v8f;
typedef __attribute__((ext_vector_type(2)))  __bf16 v2bf;
typedef __attribute__((ext_vector_type(8)))  __bf16 v8bf;
typedef __attribute__((ext_vector_type(16))) __bf16 v16bf;

__device__ __forceinline__ unsigned short f2bf_bits(float x) {
  __bf16 b = (__bf16)x;
  return __builtin_bit_cast(unsigned short, b);
}

__device__ __forceinline__ v16bf mk16(v8bf lo, v8bf hi) {
  v16bf r;
#pragma unroll
  for (int i = 0; i < 8; ++i) { r[i] = lo[i]; r[i + 8] = hi[i]; }
  return r;
}

__device__ __forceinline__ float lrelu(float x) {
  return fmaxf(x, ALPHA * x);   // x>0 ? x : 0.2x
}

// ---------------------------------------------------------------------------
// K1: Wh[8192][128] (f32) = h[8192][256] @ W[256][128]; WhT[128][8192] (bf16)
// One wave per 16x16 output tile, f32 WMMA over K in steps of 4.
// ---------------------------------------------------------------------------
__global__ __launch_bounds__(256) void gat_gemm1(
    const float* __restrict__ h, const float* __restrict__ W,
    float* __restrict__ Wh, unsigned short* __restrict__ WhT) {
  const int wave = threadIdx.x >> 5;
  const int lane = threadIdx.x & 31;
  const int tile = blockIdx.x * 8 + wave;          // 4096 tiles
  const int rowT = tile >> 3;                      // 512 row tiles
  const int colT = tile & 7;                       // 8  col tiles
  const int l16  = lane & 15;
  const int half = lane >> 4;
  const int row  = rowT * 16 + l16;
  const int col  = colT * 16 + l16;

  v8f c = {};
#pragma unroll 4
  for (int k = 0; k < INF_; k += 4) {
    const int kk = k + 2 * half;
    // A 16x4: lanes 0-15 K={0,1}, lanes 16-31 K={2,3}
    v2f aV = *(const v2f*)(h + (size_t)row * INF_ + kk);
    // B 4x16: lanes 0-15 K={0,1}, lanes 16-31 K={2,3}, N = l16
    v2f bV;
    bV[0] = W[(size_t)kk * OUTF + col];
    bV[1] = W[(size_t)(kk + 1) * OUTF + col];
    c = __builtin_amdgcn_wmma_f32_16x16x4_f32(
        false, aV, false, bV, (short)0, c, false, false);
  }

#pragma unroll
  for (int v = 0; v < 8; ++v) {
    const int r = rowT * 16 + v + 8 * half;        // C layout: M = v + 8*half
    Wh[(size_t)r * OUTF + col] = c[v];
    WhT[(size_t)col * GN + r]  = f2bf_bits(c[v]);
  }
}

// ---------------------------------------------------------------------------
// K2: wh1[i] = (Wh[i].a1)*log2e, wh2[i] = (Wh[i].a2)*log2e, block max of wh2
// ---------------------------------------------------------------------------
__global__ __launch_bounds__(256) void gat_attvec(
    const float* __restrict__ Wh, const float* __restrict__ a,
    float* __restrict__ wh1, float* __restrict__ wh2,
    float* __restrict__ partmax) {
  const int r = blockIdx.x * 256 + threadIdx.x;
  float s1 = 0.f, s2 = 0.f;
#pragma unroll
  for (int k = 0; k < OUTF; k += 4) {
    float4 v  = *(const float4*)(Wh + (size_t)r * OUTF + k);
    float4 a1 = *(const float4*)(a + k);
    float4 a2 = *(const float4*)(a + OUTF + k);
    s1 += v.x * a1.x + v.y * a1.y + v.z * a1.z + v.w * a1.w;
    s2 += v.x * a2.x + v.y * a2.y + v.z * a2.z + v.w * a2.w;
  }
  s1 *= LOG2E;                   // log2-domain (lrelu commutes with pos scale)
  s2 *= LOG2E;
  wh1[r] = s1;
  wh2[r] = s2;

  __shared__ float red[256];
  red[threadIdx.x] = s2;
  __syncthreads();
  for (int off = 128; off > 0; off >>= 1) {
    if (threadIdx.x < off)
      red[threadIdx.x] = fmaxf(red[threadIdx.x], red[threadIdx.x + off]);
    __syncthreads();
  }
  if (threadIdx.x == 0) partmax[blockIdx.x] = red[0];
}

__global__ void gat_finmax(const float* __restrict__ partmax,
                           float* __restrict__ wh2max) {
  float v = partmax[threadIdx.x];                  // 32 threads, 32 partials
#pragma unroll
  for (int off = 16; off > 0; off >>= 1)
    v = fmaxf(v, __shfl_xor(v, off, 32));
  if (threadIdx.x == 0) *wh2max = v;
}

// ---------------------------------------------------------------------------
// K3: fused masked softmax + attention @ Wh + ELU.
// 64 blocks x 256 threads (8 waves); wave w owns 16 rows. Column tiles of 64
// (two WMMA K-steps). Double-buffered LDS B tile; row sums via P @ ones WMMA.
// ---------------------------------------------------------------------------
#define TJ    64
#define BPAD  72     // bf16 elements per LDS B row (bank-clean, 16B aligned)
#define APAD  72     // bf16 elements per LDS A row

__global__ __launch_bounds__(256) void gat_fused(
    const int* __restrict__ adj, const float* __restrict__ wh1,
    const float* __restrict__ wh2, const float* __restrict__ wh2max_p,
    const unsigned short* __restrict__ WhT, float* __restrict__ out) {
  __shared__ __bf16 ldsB[2][OUTF * BPAD];          // 2 x 18432 B
  __shared__ __bf16 ldsA[8][16 * APAD];            // 18432 B (wave-private)

  const int tid  = threadIdx.x;
  const int lane = tid & 31;
  const int wave = tid >> 5;
  const int l16  = lane & 15;
  const int half = lane >> 4;                      // 0: lanes 0-15, 1: 16-31
  const int rbase = blockIdx.x * 128 + wave * 16;
  const float wh2maxL = *wh2max_p;                 // already * log2e

  // per-row constants (log2 domain), hoisted out of the tile loop
  const float w1l = wh1[rbase + l16];
  float w1L[16], cmL[16];
#pragma unroll
  for (int m = 0; m < 16; ++m) {
    const float w1m = __shfl(w1l, m, 32);
    w1L[m] = w1m;
    cmL[m] = lrelu(w1m + wh2maxL);                 // row normalizer >= row max
  }

  v16bf ones;
#pragma unroll
  for (int i = 0; i < 16; ++i) ones[i] = (__bf16)1.0f;

  v8f acc[8] = {};
  v8f accsum = {};                                 // rowsum = P @ ones

  auto loadB = [&](int j0, int buf) {              // coop: 16KB / 256 thr
    const int f = tid >> 1, part = tid & 1;        // 64B per thread
    const uint4* src = (const uint4*)(WhT + (size_t)f * GN + j0 + part * 32);
    uint4 d0 = src[0], d1 = src[1], d2 = src[2], d3 = src[3];
    uint4* dst = (uint4*)&ldsB[buf][f * BPAD + part * 32];
    dst[0] = d0; dst[1] = d1; dst[2] = d2; dst[3] = d3;
  };

  loadB(0, 0);

  __bf16* myA = ldsA[wave];
  const int ntiles = GN / TJ;                      // 128

  for (int t = 0; t < ntiles; ++t) {
    const int j0  = t * TJ;
    const int cur = t & 1;

    // per-lane wh2 for 2 columns (log2 domain)
    const v2f w2 = *(const v2f*)(wh2 + j0 + 2 * lane);

    __syncthreads();   // iter t-1 reads of buf[cur^1] + writes of buf[cur] done
    if (t + 1 < ntiles) loadB(j0 + TJ, cur ^ 1);

    // prefetch adj 4 tiles ahead (each lane prefetches its own address:
    // one global_prefetch_b8 per row covers the whole future 16x64 tile)
    if (j0 + 4 * TJ < GN) {
#pragma unroll
      for (int m = 0; m < 16; ++m)
        __builtin_prefetch(adj + (size_t)(rbase + m) * GN + j0 + 4 * TJ + 2 * lane,
                           0, 1);
    }

    int2 av[16];
#pragma unroll
    for (int m = 0; m < 16; ++m)                   // coalesced 256B per row
      av[m] = *(const int2*)(adj + (size_t)(rbase + m) * GN + j0 + 2 * lane);

#pragma unroll
    for (int m = 0; m < 16; ++m) {
      const float x0 = w1L[m] + w2[0];
      const float x1 = w1L[m] + w2[1];
      float t0 = fmaxf(x0, ALPHA * x0) - cmL[m];   // <= 0 always
      float t1 = fmaxf(x1, ALPHA * x1) - cmL[m];
      float p0 = exp2f(t0);
      float p1 = exp2f(t1);
      p0 = (av[m].x > 0) ? p0 : 0.f;
      p1 = (av[m].y > 0) ? p1 : 0.f;
      v2bf pk;                                     // v_cvt_pk_bf16_f32
      pk[0] = (__bf16)p0;
      pk[1] = (__bf16)p1;
      *(v2bf*)&myA[m * APAD + 2 * lane] = pk;      // single ds_store_b32
    }

#pragma unroll
    for (int s = 0; s < 2; ++s) {                  // two K=32 steps
      // A frag: lanes 0-15 row l16 K{0..7,16..23}; 16-31 K{8..15,24..31}
      const __bf16* arow = myA + l16 * APAD + s * 32 + half * 8;
      v16bf afrag = mk16(*(const v8bf*)(arow), *(const v8bf*)(arow + 16));
#pragma unroll
      for (int nb = 0; nb < 8; ++nb) {
        // B frag: lane n<16: N=n, K{0..7,16..23}; lanes>=16: N=n-16, K{8..15,24..31}
        const __bf16* brow = &ldsB[cur][(nb * 16 + l16) * BPAD + s * 32 + half * 8];
        v16bf bfrag = mk16(*(const v8bf*)(brow), *(const v8bf*)(brow + 16));
        acc[nb] = __builtin_amdgcn_wmma_f32_16x16x32_bf16(
            false, afrag, false, bfrag, (short)0, acc[nb], false, false);
      }
      accsum = __builtin_amdgcn_wmma_f32_16x16x32_bf16(
          false, afrag, false, ones, (short)0, accsum, false, false);
    }
  }

  // epilogue: accsum[v] == rowsum of row M=v+8*half (same lane, C layout)
#pragma unroll
  for (int nb = 0; nb < 8; ++nb) {
#pragma unroll
    for (int v = 0; v < 8; ++v) {
      float l = accsum[v];
      l = (l > 0.f) ? l : 1.f;                     // isolated-node guard
      const float x = acc[nb][v] / l;
      const float y = (x > 0.f) ? x : (exp2f(x * LOG2E) - 1.f);
      const int row = rbase + v + 8 * half;
      out[(size_t)row * OUTF + nb * 16 + l16] = y;
    }
  }
}

// ---------------------------------------------------------------------------
extern "C" void kernel_launch(void* const* d_in, const int* in_sizes, int n_in,
                              void* d_out, int out_size, void* d_ws,
                              size_t ws_size, hipStream_t stream) {
  (void)in_sizes; (void)n_in; (void)out_size; (void)ws_size;
  const float* h   = (const float*)d_in[0];   // [8192, 256]
  const int*   adj = (const int*)d_in[1];     // [8192, 8192]
  const float* W   = (const float*)d_in[2];   // [256, 128]
  const float* a   = (const float*)d_in[3];   // [256, 1]
  float* out = (float*)d_out;                 // [8192, 128]

  char* ws = (char*)d_ws;
  float* Wh           = (float*)ws;           ws += (size_t)GN * OUTF * 4;  // 4 MB
  unsigned short* WhT = (unsigned short*)ws;  ws += (size_t)GN * OUTF * 2;  // 2 MB
  float* wh1          = (float*)ws;           ws += (size_t)GN * 4;
  float* wh2          = (float*)ws;           ws += (size_t)GN * 4;
  float* partmax      = (float*)ws;           ws += 32 * 4;
  float* wh2max       = (float*)ws;

  gat_gemm1 <<<512, 256, 0, stream>>>(h, W, Wh, WhT);
  gat_attvec<<<32,  256, 0, stream>>>(Wh, a, wh1, wh2, partmax);
  gat_finmax<<<1,   32,  0, stream>>>(partmax, wh2max);
  gat_fused <<<64,  256, 0, stream>>>(adj, wh1, wh2, wh2max, WhT, out);
}
